// MultiheadAttention_16604343567114
// MI455X (gfx1250) — matrix operand order repaired and checked by reading, not compiled, not run on hardware
//
#include <hip/hip_runtime.h>
#include <hip/hip_bf16.h>

// ---------------------------------------------------------------------------
// MHA forward for MI455X (gfx1250, wave32, WMMA bf16 path)
// B=4, S=2048, D=1024, H=16, hd=64
// ---------------------------------------------------------------------------

#define B_SZ   4
#define S_LEN  2048
#define D_MOD  1024
#define N_HEAD 16
#define HD     64
#define M_ROWS (B_SZ * S_LEN)      // 8192
#define N_QKV  (3 * D_MOD)         // 3072

typedef __attribute__((ext_vector_type(8)))  __bf16 v8bf;
typedef __attribute__((ext_vector_type(16))) __bf16 v16bf;
typedef __attribute__((ext_vector_type(8)))  float  v8f;

struct bfpair { v8bf lo, hi; };

// A-operand fragment (16x32, 16-bit): lane r=lane&15 holds row r.
// lanes 0-15:  K = {k0..k0+7} U {k0+16..k0+23}, k0 = 0
// lanes 16-31: same with k0 = 8
// p must already point at (row_base + kk + k0).
static __device__ __forceinline__ v16bf load_frag_a(const __hip_bfloat16* p) {
    bfpair pr;
    pr.lo = *reinterpret_cast<const v8bf*>(p);
    pr.hi = *reinterpret_cast<const v8bf*>(p + 16);
    return __builtin_bit_cast(v16bf, pr);
}

// B-operand fragment (32x16, 16-bit): lane holds column N=lane&15,
// K = {kstart..kstart+7} U {kstart+16..kstart+23}, kstart = 8*(lane>>4).
// p must already point at (col_base + kk + kstart).
static __device__ __forceinline__ v16bf load_frag_b(const __hip_bfloat16* p) {
    bfpair pr;
    pr.lo = *reinterpret_cast<const v8bf*>(p);
    pr.hi = *reinterpret_cast<const v8bf*>(p + 16);
    return __builtin_bit_cast(v16bf, pr);
}

static __device__ __forceinline__ v8f wmma_bf16(v16bf a, v16bf b, v8f c) {
    return __builtin_amdgcn_wmma_f32_16x16x32_bf16(
        false, a, false, b, (short)0, c, false, false);
}

static __device__ __forceinline__ float redmax16(float v) {
    v = fmaxf(v, __shfl_xor(v, 1));
    v = fmaxf(v, __shfl_xor(v, 2));
    v = fmaxf(v, __shfl_xor(v, 4));
    v = fmaxf(v, __shfl_xor(v, 8));
    return v;
}
static __device__ __forceinline__ float redsum16(float v) {
    v += __shfl_xor(v, 1);
    v += __shfl_xor(v, 2);
    v += __shfl_xor(v, 4);
    v += __shfl_xor(v, 8);
    return v;
}

// ---------------------------------------------------------------------------
// fp32 -> bf16 cast
// ---------------------------------------------------------------------------
__global__ void cast_f32_bf16(const float* __restrict__ in,
                              __hip_bfloat16* __restrict__ out, int n) {
    int i = blockIdx.x * blockDim.x + threadIdx.x;
    if (i < n) out[i] = __float2bfloat16(in[i]);
}

// ---------------------------------------------------------------------------
// Shared GEMM mainloop: wave computes 32x64 tile of C = A[M,K] * W[N,K]^T
// ---------------------------------------------------------------------------
static __device__ __forceinline__ void gemm_mainloop(
    const __hip_bfloat16* __restrict__ A,
    const __hip_bfloat16* __restrict__ W,
    int K, int m0, int n0, int lane, v8f acc[2][4]) {

    const int r   = lane & 15;
    const int k0a = (lane >> 4) << 3;   // A fragment half offset
    const int k0b = (lane >> 4) << 3;   // B fragment half offset

    const __hip_bfloat16* a0 = A + (size_t)(m0 + r)      * K + k0a;
    const __hip_bfloat16* a1 = A + (size_t)(m0 + 16 + r) * K + k0a;
    const __hip_bfloat16* b0 = W + (size_t)(n0 + r)      * K + k0b;
    const __hip_bfloat16* b1 = W + (size_t)(n0 + 16 + r) * K + k0b;
    const __hip_bfloat16* b2 = W + (size_t)(n0 + 32 + r) * K + k0b;
    const __hip_bfloat16* b3 = W + (size_t)(n0 + 48 + r) * K + k0b;

#pragma unroll 2
    for (int kk = 0; kk < K; kk += 32) {
        v16bf af0 = load_frag_a(a0 + kk);
        v16bf af1 = load_frag_a(a1 + kk);
        v16bf bf0 = load_frag_b(b0 + kk);
        v16bf bf1 = load_frag_b(b1 + kk);
        v16bf bf2 = load_frag_b(b2 + kk);
        v16bf bf3 = load_frag_b(b3 + kk);
        acc[0][0] = wmma_bf16(af0, bf0, acc[0][0]);
        acc[0][1] = wmma_bf16(af0, bf1, acc[0][1]);
        acc[0][2] = wmma_bf16(af0, bf2, acc[0][2]);
        acc[0][3] = wmma_bf16(af0, bf3, acc[0][3]);
        acc[1][0] = wmma_bf16(af1, bf0, acc[1][0]);
        acc[1][1] = wmma_bf16(af1, bf1, acc[1][1]);
        acc[1][2] = wmma_bf16(af1, bf2, acc[1][2]);
        acc[1][3] = wmma_bf16(af1, bf3, acc[1][3]);
    }
}

// ---------------------------------------------------------------------------
// QKV projection: [8192,1024] x [3072,1024]^T, scatter epilogue into
//   q,k : [B,H,S,hd] bf16      v : [B,H,hd,S] bf16 (transposed for P@V)
// grid (M/64, N/256), block 256 (8 waves, 2x4 wave grid)
// ---------------------------------------------------------------------------
__global__ void gemm_qkv(const __hip_bfloat16* __restrict__ A,
                         const __hip_bfloat16* __restrict__ W,
                         __hip_bfloat16* __restrict__ qo,
                         __hip_bfloat16* __restrict__ ko,
                         __hip_bfloat16* __restrict__ vto) {
    const int lane = threadIdx.x & 31;
    const int wid  = threadIdx.x >> 5;
    const int m0   = blockIdx.x * 64  + (wid >> 2) * 32;
    const int n0   = blockIdx.y * 256 + (wid & 3) * 64;

    v8f acc[2][4];
#pragma unroll
    for (int i = 0; i < 2; ++i)
#pragma unroll
        for (int j = 0; j < 4; ++j) { v8f z = {}; acc[i][j] = z; }

    gemm_mainloop(A, W, D_MOD, m0, n0, lane, acc);

    const int halfoff = (lane >> 4) << 3;
#pragma unroll
    for (int i = 0; i < 2; ++i) {
#pragma unroll
        for (int j = 0; j < 4; ++j) {
#pragma unroll
            for (int t = 0; t < 8; ++t) {
                int mr = m0 + i * 16 + t + halfoff;
                int nc = n0 + j * 16 + (lane & 15);
                int bb = mr >> 11;           // / S_LEN
                int ss = mr & (S_LEN - 1);
                int which = nc >> 10;        // / D_MOD
                int rem   = nc & (D_MOD - 1);
                int hh = rem >> 6;
                int dd = rem & (HD - 1);
                __hip_bfloat16 bv = __float2bfloat16(acc[i][j][t]);
                if (which == 0)
                    qo[(((size_t)(bb * N_HEAD + hh) * S_LEN) + ss) * HD + dd] = bv;
                else if (which == 1)
                    ko[(((size_t)(bb * N_HEAD + hh) * S_LEN) + ss) * HD + dd] = bv;
                else
                    vto[(((size_t)(bb * N_HEAD + hh) * HD) + dd) * S_LEN + ss] = bv;
            }
        }
    }
}

// ---------------------------------------------------------------------------
// Flash attention: one wave per (b, h, 16-query tile). Online softmax,
// causal mask, P routed through LDS (C-layout -> A-fragment layout).
// block 256 (8 waves), grid 1024.
// ---------------------------------------------------------------------------
__global__ void flash_attn(const __hip_bfloat16* __restrict__ q,
                           const __hip_bfloat16* __restrict__ k,
                           const __hip_bfloat16* __restrict__ vt,
                           __hip_bfloat16* __restrict__ ctx) {
    __shared__ __align__(16) __hip_bfloat16 sP[8][16 * 32];

    const int lane = threadIdx.x & 31;
    const int wid  = threadIdx.x >> 5;
    const int w    = blockIdx.x * 8 + wid;   // 0..8191
    const int qt   = w & 127;                // S/16 tiles
    const int hh   = (w >> 7) & (N_HEAD - 1);
    const int bb   = w >> 11;
    const int q0   = qt * 16;

    const int r       = lane & 15;
    const int k0a     = (lane >> 4) << 3;
    const int k0b     = (lane >> 4) << 3;
    const int halfoff = (lane >> 4) << 3;

    const __hip_bfloat16* Qb = q  + ((size_t)(bb * N_HEAD + hh) * S_LEN + q0) * HD;
    const __hip_bfloat16* Kb = k  + ((size_t)(bb * N_HEAD + hh) * S_LEN) * HD;
    const __hip_bfloat16* Vb = vt + ((size_t)(bb * N_HEAD + hh) * HD) * S_LEN;
    __hip_bfloat16* myP = &sP[wid][0];

    // Q fragments (A operand): rows q0..q0+15, d split into two 32-chunks
    v16bf qa0 = load_frag_a(Qb + (size_t)r * HD + k0a);
    v16bf qa1 = load_frag_a(Qb + (size_t)r * HD + 32 + k0a);

    v8f oacc[4];
#pragma unroll
    for (int o = 0; o < 4; ++o) { v8f z = {}; oacc[o] = z; }
    float mrun[8], lrun[8];
#pragma unroll
    for (int t = 0; t < 8; ++t) { mrun[t] = -1e30f; lrun[t] = 0.0f; }

    const float scale = 0.125f;   // 1/sqrt(64)
    const int kend = q0 + 16;

    for (int kb = 0; kb < kend; kb += 32) {
        // S = Q K^T : K rows act as B columns (contiguous over d)
        const __hip_bfloat16* kr0 = Kb + (size_t)(kb + r) * HD + k0b;
        const __hip_bfloat16* kr1 = Kb + (size_t)(kb + 16 + r) * HD + k0b;
        v16bf kf00 = load_frag_b(kr0);        // keys kb..kb+15, d 0..31
        v16bf kf01 = load_frag_b(kr0 + 32);   // d 32..63
        v16bf kf10 = load_frag_b(kr1);        // keys kb+16..kb+31
        v16bf kf11 = load_frag_b(kr1 + 32);

        v8f s0 = {}, s1 = {};
        s0 = wmma_bf16(qa0, kf00, s0);
        s0 = wmma_bf16(qa1, kf01, s0);
        s1 = wmma_bf16(qa0, kf10, s1);
        s1 = wmma_bf16(qa1, kf11, s1);

        const int key0 = kb + (lane & 15);
        const int key1 = key0 + 16;
        float alpha[8];
#pragma unroll
        for (int t = 0; t < 8; ++t) {
            int qrow = q0 + t + halfoff;
            float v0 = s0[t] * scale;
            float v1 = s1[t] * scale;
            if (key0 > qrow) v0 = -1e30f;
            if (key1 > qrow) v1 = -1e30f;
            float mx   = redmax16(fmaxf(v0, v1));
            float mnew = fmaxf(mrun[t], mx);
            float a    = __expf(mrun[t] - mnew);
            float p0   = __expf(v0 - mnew);
            float p1   = __expf(v1 - mnew);
            float rs   = redsum16(p0 + p1);
            lrun[t] = lrun[t] * a + rs;
            mrun[t] = mnew;
            alpha[t] = a;
            int rl = t + halfoff;
            myP[rl * 32 + (lane & 15)]      = __float2bfloat16(p0);
            myP[rl * 32 + (lane & 15) + 16] = __float2bfloat16(p1);
        }
        // rescale running output
#pragma unroll
        for (int o = 0; o < 4; ++o)
#pragma unroll
            for (int t = 0; t < 8; ++t) oacc[o][t] *= alpha[t];

        // same-wave LDS RAW: DS ops are in-order per wave; fence for safety
        asm volatile("s_wait_dscnt 0" ::: "memory");

        // P fragment (A operand) from LDS, V fragments (B operand) from Vt
        v16bf pa = load_frag_a(myP + r * 32 + k0a);
#pragma unroll
        for (int o = 0; o < 4; ++o) {
            v16bf vf = load_frag_b(Vb + (size_t)(o * 16 + r) * S_LEN + kb + k0b);
            oacc[o] = wmma_bf16(pa, vf, oacc[o]);
        }
    }

    // normalize and store ctx in [B,S,D] bf16 for the output projection
    float inv[8];
#pragma unroll
    for (int t = 0; t < 8; ++t) inv[t] = 1.0f / lrun[t];
#pragma unroll
    for (int o = 0; o < 4; ++o) {
#pragma unroll
        for (int t = 0; t < 8; ++t) {
            int qrow = q0 + t + halfoff;
            int dd   = o * 16 + (lane & 15);
            ctx[((size_t)(bb * S_LEN + qrow)) * D_MOD + hh * HD + dd] =
                __float2bfloat16(oacc[o][t] * inv[t]);
        }
    }
}

// ---------------------------------------------------------------------------
// Output projection: ctx[8192,1024] x w_proj[1024,1024]^T + bias -> fp32 out
// grid (M/64, N/256), block 256
// ---------------------------------------------------------------------------
__global__ void gemm_proj(const __hip_bfloat16* __restrict__ A,
                          const __hip_bfloat16* __restrict__ W,
                          const float* __restrict__ bias,
                          float* __restrict__ out) {
    const int lane = threadIdx.x & 31;
    const int wid  = threadIdx.x >> 5;
    const int m0   = blockIdx.x * 64  + (wid >> 2) * 32;
    const int n0   = blockIdx.y * 256 + (wid & 3) * 64;

    v8f acc[2][4];
#pragma unroll
    for (int i = 0; i < 2; ++i)
#pragma unroll
        for (int j = 0; j < 4; ++j) { v8f z = {}; acc[i][j] = z; }

    gemm_mainloop(A, W, D_MOD, m0, n0, lane, acc);

    const int halfoff = (lane >> 4) << 3;
#pragma unroll
    for (int i = 0; i < 2; ++i) {
#pragma unroll
        for (int j = 0; j < 4; ++j) {
            int nc = n0 + j * 16 + (lane & 15);
            float bv = bias[nc];
#pragma unroll
            for (int t = 0; t < 8; ++t) {
                int mr = m0 + i * 16 + t + halfoff;
                out[(size_t)mr * D_MOD + nc] = acc[i][j][t] + bv;
            }
        }
    }
}

// ---------------------------------------------------------------------------
// Launch
// ---------------------------------------------------------------------------
extern "C" void kernel_launch(void* const* d_in, const int* in_sizes, int n_in,
                              void* d_out, int out_size, void* d_ws, size_t ws_size,
                              hipStream_t stream) {
    (void)in_sizes; (void)n_in; (void)out_size; (void)ws_size;

    const float* x      = (const float*)d_in[0];   // [4,2048,1024]
    const float* w_qkv  = (const float*)d_in[1];   // [3072,1024]
    const float* w_proj = (const float*)d_in[2];   // [1024,1024]
    const float* b_proj = (const float*)d_in[3];   // [1024]
    float* out = (float*)d_out;

    char* ws = (char*)d_ws;
    size_t off = 0;
    __hip_bfloat16* xh  = (__hip_bfloat16*)(ws + off); off += (size_t)M_ROWS * D_MOD * 2;  // 16 MB
    __hip_bfloat16* wqh = (__hip_bfloat16*)(ws + off); off += (size_t)N_QKV  * D_MOD * 2;  //  6 MB
    __hip_bfloat16* wph = (__hip_bfloat16*)(ws + off); off += (size_t)D_MOD  * D_MOD * 2;  //  2 MB
    __hip_bfloat16* qh  = (__hip_bfloat16*)(ws + off); off += (size_t)M_ROWS * D_MOD * 2;  // 16 MB
    __hip_bfloat16* kh  = (__hip_bfloat16*)(ws + off); off += (size_t)M_ROWS * D_MOD * 2;  // 16 MB
    __hip_bfloat16* vth = (__hip_bfloat16*)(ws + off); off += (size_t)M_ROWS * D_MOD * 2;  // 16 MB
    __hip_bfloat16* cth = (__hip_bfloat16*)(ws + off); off += (size_t)M_ROWS * D_MOD * 2;  // 16 MB

    const int nx = M_ROWS * D_MOD;
    const int nq = N_QKV * D_MOD;
    const int np = D_MOD * D_MOD;
    cast_f32_bf16<<<(nx + 255) / 256, 256, 0, stream>>>(x, xh, nx);
    cast_f32_bf16<<<(nq + 255) / 256, 256, 0, stream>>>(w_qkv, wqh, nq);
    cast_f32_bf16<<<(np + 255) / 256, 256, 0, stream>>>(w_proj, wph, np);

    gemm_qkv<<<dim3(M_ROWS / 64, N_QKV / 256), 256, 0, stream>>>(xh, wqh, qh, kh, vth);

    flash_attn<<<1024, 256, 0, stream>>>(qh, kh, vth, cth);

    gemm_proj<<<dim3(M_ROWS / 64, D_MOD / 256), 256, 0, stream>>>(cth, wph, b_proj, out);
}
